// SelfAttention_70119636074780
// MI455X (gfx1250) — matrix operand order repaired and checked by reading, not compiled, not run on hardware
//
#include <hip/hip_runtime.h>
#include <hip/hip_bf16.h>

typedef __attribute__((ext_vector_type(16))) __bf16 v16bf;
typedef __attribute__((ext_vector_type(8)))  __bf16 v8bf;
typedef __attribute__((ext_vector_type(8)))  float  v8f;

#define BB 4
#define CC 256
#define CR 32
#define NN 4096
#define LOG2E 1.44269504088896340736f

// ---------------------------------------------------------------------------
// Projections: bf16 Q (B,N,Cr) PRE-SCALED by log2(e), K (B,Cr,N), V (B,N,C).
// Memory-bound; weights indexed uniformly -> scalar loads from L2.
// ---------------------------------------------------------------------------
__global__ __launch_bounds__(256) void proj_qk_kernel(
    const float* __restrict__ x,
    const float* __restrict__ wq, const float* __restrict__ bq,
    const float* __restrict__ wk, const float* __restrict__ bk,
    __bf16* __restrict__ Qo /*[B][N][CR]*/,
    __bf16* __restrict__ Ko /*[B][CR][N]*/)
{
    const int b = blockIdx.y;
    const int n = blockIdx.x * 256 + threadIdx.x;

    float qa[CR], ka[CR];
    #pragma unroll
    for (int o = 0; o < CR; ++o) { qa[o] = 0.f; ka[o] = 0.f; }

    const float* xb = x + (size_t)b * CC * NN + n;
    for (int c = 0; c < CC; ++c) {
        float xv = xb[(size_t)c * NN];
        #pragma unroll
        for (int o = 0; o < CR; ++o) {
            qa[o] = fmaf(wq[o * CC + c], xv, qa[o]);
            ka[o] = fmaf(wk[o * CC + c], xv, ka[o]);
        }
    }
    __bf16* qrow = Qo + ((size_t)b * NN + n) * CR;
    #pragma unroll
    for (int o = 0; o < CR; ++o)
        qrow[o] = (__bf16)((qa[o] + bq[o]) * LOG2E);   // fold log2(e) into Q
    #pragma unroll
    for (int o = 0; o < CR; ++o)
        Ko[((size_t)b * CR + o) * NN + n] = (__bf16)(ka[o] + bk[o]);
}

__global__ __launch_bounds__(256) void proj_v_kernel(
    const float* __restrict__ x,
    const float* __restrict__ wv, const float* __restrict__ bv,
    __bf16* __restrict__ Vo /*[B][N][CC]*/)
{
    const int b  = blockIdx.y;
    const int n  = blockIdx.x * 256 + threadIdx.x;
    const int o0 = blockIdx.z * 32;

    float va[32];
    #pragma unroll
    for (int j = 0; j < 32; ++j) va[j] = 0.f;

    const float* xb = x + (size_t)b * CC * NN + n;
    for (int c = 0; c < CC; ++c) {
        float xv = xb[(size_t)c * NN];
        #pragma unroll
        for (int j = 0; j < 32; ++j)
            va[j] = fmaf(wv[(o0 + j) * CC + c], xv, va[j]);
    }
    __bf16* vrow = Vo + ((size_t)b * NN + n) * CC + o0;
    #pragma unroll
    for (int j = 0; j < 32; ++j) vrow[j] = (__bf16)(va[j] + bv[o0 + j]);
}

// ---------------------------------------------------------------------------
// Fused flash attention. Block = (b, 64 n-rows), 8 waves of 32.
// wave -> 16 n-rows (wave&3) x 128 channels (wave>>2). m swept in 64s.
// Softmax in exp2 domain (Q pre-scaled). Row sums via WMMA against ones.
// ---------------------------------------------------------------------------
__global__ __launch_bounds__(256) void attn_kernel(
    const float* __restrict__ x,
    const __bf16* __restrict__ Q,   // [B][N][CR], scaled by log2(e)
    const __bf16* __restrict__ K,   // [B][CR][N]
    const __bf16* __restrict__ V,   // [B][N][CC]
    float* __restrict__ out)        // [B][CC][N]
{
    __shared__ __bf16 Plds[8][16][64];   // per-wave private P staging (16 KB)

    const int b    = blockIdx.y;
    const int tid  = threadIdx.x;
    const int wave = tid >> 5;
    const int lane = tid & 31;
    const int half = lane >> 4;          // 0: lanes 0-15, 1: lanes 16-31
    const int l15  = lane & 15;
    const int nbase = blockIdx.x * 64 + (wave & 3) * 16;
    const int c0    = (wave >> 2) * 128;

    // A-operand (Q): lane holds row n = nbase+l15; K-chunks {0..7,16..23} or
    // {8..15,24..31} depending on half (16-bit A 16x32 VGPR layout).
    v16bf qA;
    {
        const __bf16* qrow = Q + ((size_t)b * NN + nbase + l15) * CR + half * 8;
        v8bf lo = *(const v8bf*)(qrow);
        v8bf hi = *(const v8bf*)(qrow + 16);
        #pragma unroll
        for (int i = 0; i < 8; ++i) { qA[i] = lo[i]; qA[i + 8] = hi[i]; }
    }

    v16bf onesB;                          // all-ones B operand for row sums
    #pragma unroll
    for (int i = 0; i < 16; ++i) onesB[i] = (__bf16)1.0f;

    v8f O[8];
    #pragma unroll
    for (int t = 0; t < 8; ++t) O[t] = (v8f){};
    float mrow[8], lrow[8];
    #pragma unroll
    for (int r = 0; r < 8; ++r) { mrow[r] = -INFINITY; lrow[r] = 0.f; }

    const __bf16* Kb = K + ((size_t)b * CR + lane) * NN;  // B-operand: lane = c
    const __bf16* Vb = V + (size_t)b * NN * CC;
    const v8f z = {};

    for (int m0 = 0; m0 < NN; m0 += 64) {
        // ---- S = Q^T K : four 16x16 logit tiles (already log2-scaled) ----
        v8f s[4];
        #pragma unroll
        for (int t = 0; t < 4; ++t) {
            v16bf kb;
            #pragma unroll
            for (int i = 0; i < 16; ++i)
                kb[i] = K[((size_t)b * CR + 8 * half + (i & 7) + 16 * (i >> 3)) * NN + m0 + 16 * t + l15];
            s[t] = __builtin_amdgcn_wmma_f32_16x16x32_bf16(false, qA, false, kb,
                                                           (short)0, z, false, false);
        }
        if (m0 + 64 < NN) {
            __builtin_prefetch(Kb + m0 + 64, 0, 0);
            __builtin_prefetch(Vb + (size_t)(m0 + 64 + lane) * CC + c0, 0, 0);
            __builtin_prefetch(Vb + (size_t)(m0 + 96 + lane) * CC + c0, 0, 0);
        }

        // ---- online softmax: row max (shuffle), exp2, rescale ----
        float alpha[8];
        #pragma unroll
        for (int r = 0; r < 8; ++r) {
            float t = fmaxf(fmaxf(s[0][r], s[1][r]), fmaxf(s[2][r], s[3][r]));
            #pragma unroll
            for (int d = 1; d < 16; d <<= 1) t = fmaxf(t, __shfl_xor(t, d, 32));
            float mnew = fmaxf(mrow[r], t);
            alpha[r] = exp2f(mrow[r] - mnew);
            mrow[r] = mnew;
            #pragma unroll
            for (int tt = 0; tt < 4; ++tt) s[tt][r] = exp2f(s[tt][r] - mnew);
            #pragma unroll
            for (int tt = 0; tt < 8; ++tt) O[tt][r] *= alpha[r];
        }

        // ---- stage P (bf16) to per-wave LDS tile; D-layout -> A-layout ----
        #pragma unroll
        for (int r = 0; r < 8; ++r) {
            const int row = r + 8 * half;
            #pragma unroll
            for (int tt = 0; tt < 4; ++tt)
                Plds[wave][row][tt * 16 + l15] = (__bf16)s[tt][r];
        }
        __builtin_amdgcn_wave_barrier();   // DS in-order within wave
        v16bf pA0, pA1;
        {
            const __bf16* pr = &Plds[wave][l15][half * 8];
            v8bf a = *(const v8bf*)(pr);         // m  base..base+7
            v8bf bm = *(const v8bf*)(pr + 16);   // m  base+16..base+23
            v8bf c = *(const v8bf*)(pr + 32);    // m  32+base..
            v8bf d = *(const v8bf*)(pr + 48);    // m  48+base..
            #pragma unroll
            for (int i = 0; i < 8; ++i) {
                pA0[i] = a[i]; pA0[i + 8] = bm[i];
                pA1[i] = c[i]; pA1[i + 8] = d[i];
            }
        }
        __builtin_amdgcn_wave_barrier();

        // ---- row sums of P via WMMA against ones (replaces shuffle-reduce) ----
        v8f sumT = __builtin_amdgcn_wmma_f32_16x16x32_bf16(false, pA0, false, onesB,
                                                           (short)0, z, false, false);
        sumT = __builtin_amdgcn_wmma_f32_16x16x32_bf16(false, pA1, false, onesB,
                                                       (short)0, sumT, false, false);
        #pragma unroll
        for (int r = 0; r < 8; ++r) lrow[r] = lrow[r] * alpha[r] + sumT[r];

        // ---- O += P * V over 128 channels (8 x 16c tiles, 2 m-chunks each) ----
        const __bf16* vr0 = Vb + (size_t)(m0 + 8 * half) * CC + c0 + l15;        // m-chunk 0
        const __bf16* vr1 = Vb + (size_t)(m0 + 32 + 8 * half) * CC + c0 + l15;   // m-chunk 1
        #pragma unroll
        for (int ct = 0; ct < 8; ++ct) {
            v16bf vb0, vb1;
            #pragma unroll
            for (int i = 0; i < 16; ++i) {
                vb0[i] = vr0[(size_t)((i & 7) + 16 * (i >> 3)) * CC + 16 * ct];
                vb1[i] = vr1[(size_t)((i & 7) + 16 * (i >> 3)) * CC + 16 * ct];
            }
            O[ct] = __builtin_amdgcn_wmma_f32_16x16x32_bf16(false, pA0, false, vb0,
                                                            (short)0, O[ct], false, false);

            O[ct] = __builtin_amdgcn_wmma_f32_16x16x32_bf16(false, pA1, false, vb1,
                                                            (short)0, O[ct], false, false);
        }
    }

    // ---- epilogue: normalize, add residual, scatter to (B,C,N) f32 ----
    #pragma unroll
    for (int r = 0; r < 8; ++r) {
        const float inv = 1.0f / lrow[r];
        const int n = nbase + r + 8 * half;
        size_t idx = ((size_t)b * CC + c0 + l15) * (size_t)NN + n;
        #pragma unroll
        for (int ct = 0; ct < 8; ++ct) {
            size_t k = idx + (size_t)ct * 16 * NN;
            out[k] = O[ct][r] * inv + x[k];
        }
    }
}

// ---------------------------------------------------------------------------
extern "C" void kernel_launch(void* const* d_in, const int* in_sizes, int n_in,
                              void* d_out, int out_size, void* d_ws, size_t ws_size,
                              hipStream_t stream) {
    const float* x  = (const float*)d_in[0];
    const float* wq = (const float*)d_in[1];
    const float* bq = (const float*)d_in[2];
    const float* wk = (const float*)d_in[3];
    const float* bk = (const float*)d_in[4];
    const float* wv = (const float*)d_in[5];
    const float* bv = (const float*)d_in[6];
    float* out = (float*)d_out;

    // Workspace: Q (1 MB) | K (1 MB) | V (8 MB), all bf16.
    __bf16* Qw = (__bf16*)d_ws;
    __bf16* Kw = Qw + (size_t)BB * NN * CR;
    __bf16* Vw = Kw + (size_t)BB * CR * NN;

    proj_qk_kernel<<<dim3(NN / 256, BB), 256, 0, stream>>>(x, wq, bq, wk, bk, Qw, Kw);
    proj_v_kernel<<<dim3(NN / 256, BB, CC / 32), 256, 0, stream>>>(x, wv, bv, Vw);
    attn_kernel<<<dim3(NN / 64, BB), 256, 0, stream>>>(x, Qw, Kw, Vw, out);
}
